// EnsembleModel_61718680044080
// MI455X (gfx1250) — compile-verified, matrix operand order checked by reading, not run on hardware
//
#include <hip/hip_runtime.h>

// ---------------------------------------------------------------------------
// Types for gfx1250 WMMA / TDM
// ---------------------------------------------------------------------------
typedef __bf16 bf16;
typedef __attribute__((ext_vector_type(16))) __bf16   v16bf;
typedef __attribute__((ext_vector_type(8)))  float    v8f;
typedef __attribute__((ext_vector_type(4)))  unsigned v4u;
typedef __attribute__((ext_vector_type(8)))  int      v8i;
typedef __attribute__((ext_vector_type(4)))  int      v4i;

__device__ __forceinline__ bf16 f2bf(float f) {
  union { float f; unsigned u; } v; v.f = f;
  unsigned r = v.u + 0x7FFFu + ((v.u >> 16) & 1u);   // round-to-nearest-even
  unsigned short h = (unsigned short)(r >> 16);
  bf16 b; __builtin_memcpy(&b, &h, 2);
  return b;
}

// ---------------------------------------------------------------------------
// fp32 -> bf16 conversion (grid-stride)
// ---------------------------------------------------------------------------
__global__ void f32_to_bf16_kernel(const float* __restrict__ src,
                                   bf16* __restrict__ dst, int n) {
  for (int i = blockIdx.x * blockDim.x + threadIdx.x; i < n;
       i += gridDim.x * blockDim.x)
    dst[i] = f2bf(src[i]);
}

// ---------------------------------------------------------------------------
// Tensor Data Mover: 2D tile load Global -> LDS (D# per CDNA5 ISA)
//   - data_size = 2 bytes (bf16)
//   - LDS padding: 4 DWORDs (16B) after every 16 DWORDs (64B row) => row
//     stride 80B, matching the [.][LDT=40] shared tiles below.
//   - tensor_dim = remaining extent (hardware zero-fills OOB reads)
// 6-arg builtin: (uint32x4 g0, int32x8 g1, int32x4, int32x4, int32x8, i32 cpol)
// ---------------------------------------------------------------------------
__device__ __forceinline__ void tdm_load_tile_2d(unsigned lds_byte_addr,
                                                 const bf16* gaddr,
                                                 unsigned dim0_rem,   // elems
                                                 unsigned dim1_rem,   // rows
                                                 unsigned long long stride0, // elems
                                                 unsigned tile0, unsigned tile1) {
  const unsigned long long ga = (unsigned long long)(uintptr_t)gaddr;
  v4u g0;
  g0[0] = 1u;                                            // count=1, user D#
  g0[1] = lds_byte_addr;                                 // lds_addr
  g0[2] = (unsigned)(ga & 0xFFFFFFFFu);                  // global_addr[31:0]
  g0[3] = (unsigned)((ga >> 32) & 0x01FFFFFFu)           // global_addr[56:32]
        | (2u << 30);                                    // type = 2 ("image")
  v8i g1;
  g1[0] = (int)((1u << 16)      // data_size = 1 -> 2 bytes
              | (1u << 20)      // pad_enable
              | (3u << 22)      // pad_interval: 16 DWORDs (64B) between pads
              | (3u << 25));    // pad_amount:   4 DWORDs (16B)
  g1[1] = (int)((dim0_rem & 0xFFFFu) << 16);             // tensor_dim0[15:0]
  g1[2] = (int)(((dim0_rem >> 16) & 0xFFFFu)
              | ((dim1_rem & 0xFFFFu) << 16));           // dim0 hi | dim1 lo
  g1[3] = (int)(((dim1_rem >> 16) & 0xFFFFu)
              | ((tile0 & 0xFFFFu) << 16));              // dim1 hi | tile_dim0
  g1[4] = (int)(tile1 & 0xFFFFu);                        // tile_dim1 (tile_dim2=0)
  g1[5] = (int)(unsigned)(stride0 & 0xFFFFFFFFu);        // tensor_dim0_stride lo
  g1[6] = (int)(unsigned)((stride0 >> 32) & 0xFFFFu);    // stride hi (stride1=0)
  g1[7] = 0;
  const v4i z4 = {0, 0, 0, 0};                           // groups 2/3 unused (2D)
  const v8i z8 = {0, 0, 0, 0, 0, 0, 0, 0};
  __builtin_amdgcn_tensor_load_to_lds(g0, g1, z4, z4, z8, 0);
}

// ---------------------------------------------------------------------------
// WMMA bf16 GEMM:  C[M,N] = A[M,K] * B
//   B_NMAJOR = true : B stored [N,K] row-major (computes A * B^T)
//   B_NMAJOR = false: B stored [K,N] row-major
// Block tile 128x128, K-step 32 (v_wmma_f32_16x16x32_bf16).
// 256 threads = 8 wave32; wave grid 4(rows) x 2(cols); each wave 32x64.
// Double-buffered LDS; A tile (and B in NT mode) staged asynchronously by the
// Tensor Data Mover under the WMMAs; one barrier per K-step.
// ---------------------------------------------------------------------------
#define BM 128
#define BN 128
#define BKT 32
#define LDT 40   // padded LDS row stride (elements) == TDM pad layout

// 16-bit A/B fragment: lane={hi,idx}; VGPR pair j: k = 2j+8*hi (j<4), else
// 16 + 2(j-4) + 8*hi. Contiguous 8-element runs -> compiler emits ds_load_b128.
__device__ __forceinline__ v16bf load_frag_lds(const bf16* rowbase, int lane) {
  const int r  = lane & 15;
  const int hi = (lane >> 4) * 8;
  const bf16* p = rowbase + r * LDT;
  v16bf f;
#pragma unroll
  for (int j = 0; j < 4; ++j) {
    int k = 2 * j + hi;
    f[2 * j]     = p[k];
    f[2 * j + 1] = p[k + 1];
  }
#pragma unroll
  for (int j = 0; j < 4; ++j) {
    int k = 16 + 2 * j + hi;
    f[8 + 2 * j] = p[k];
    f[9 + 2 * j] = p[k + 1];
  }
  return f;
}

template <bool B_NMAJOR, bool OUT_BF16>
__global__ __launch_bounds__(256) void gemm_bf16_wmma(
    const bf16* __restrict__ A, const bf16* __restrict__ B,
    void* __restrict__ Cout, int M, int N, int Kd) {
  __shared__ bf16 As[2][BM][LDT];
  __shared__ bf16 Bs[2][BN][LDT];   // stored [n][k]

  const int tid  = threadIdx.x;
  const int lane = tid & 31;
  const int wave = tid >> 5;
  const int wm   = wave & 3;     // row group: 32 rows each
  const int wn   = wave >> 2;    // col group: 64 cols each
  const int m0   = blockIdx.y * BM;
  const int n0   = blockIdx.x * BN;

  const bf16 zbf = f2bf(0.0f);

  v8f acc[2][4];
#pragma unroll
  for (int i = 0; i < 2; ++i)
#pragma unroll
    for (int j = 0; j < 4; ++j)
      acc[i][j] = (v8f){0.f, 0.f, 0.f, 0.f, 0.f, 0.f, 0.f, 0.f};

  const int ktiles = (Kd + BKT - 1) / BKT;

  // --- staging helpers -----------------------------------------------------
  auto stage_tdm = [&](int kt, int buf) {     // A always; B too in NT mode
    const int k0 = kt * BKT;
    tdm_load_tile_2d((unsigned)(uintptr_t)&As[buf][0][0],
                     A + (size_t)m0 * Kd + k0,
                     (unsigned)(Kd - k0), (unsigned)(M - m0),
                     (unsigned long long)Kd, BKT, BM);
    if constexpr (B_NMAJOR) {
      tdm_load_tile_2d((unsigned)(uintptr_t)&Bs[buf][0][0],
                       B + (size_t)n0 * Kd + k0,
                       (unsigned)(Kd - k0), (unsigned)(N - n0),
                       (unsigned long long)Kd, BKT, BN);
    }
  };
  auto stage_B_manual = [&](int kt, int buf) {  // NN mode: transpose into LDS
    const int k0 = kt * BKT;
    const int k  = tid >> 3;
    const int nc = (tid & 7) * 16;
    const int gk = k0 + k;
    const int gn = n0 + nc;
    bf16 tmp[16];
    if (gk < Kd && gn + 16 <= N) {
      *(uint4*)tmp = *(const uint4*)(B + (size_t)gk * N + gn);
    } else {
#pragma unroll
      for (int e = 0; e < 16; ++e)
        tmp[e] = (gk < Kd && gn + e < N) ? B[(size_t)gk * N + gn + e] : zbf;
    }
#pragma unroll
    for (int e = 0; e < 16; ++e) Bs[buf][nc + e][k] = tmp[e];
    // prefetch the tile after next (lowers to global_prefetch_b8)
    if (gk + BKT < Kd)
      __builtin_prefetch(B + (size_t)(gk + BKT) * N + gn, 0, 1);
  };

  // --- prologue: stage tile 0 into buffer 0 --------------------------------
  if (wave == 0) {
    stage_tdm(0, 0);
    __builtin_amdgcn_s_wait_tensorcnt(0);
  }
  if constexpr (!B_NMAJOR) stage_B_manual(0, 0);
  __syncthreads();

  // --- main loop: DMA tile kt+1 under the WMMAs on tile kt ------------------
  for (int kt = 0; kt < ktiles; ++kt) {
    const int cur = kt & 1;
    const int nxt = cur ^ 1;
    const bool have_next = (kt + 1 < ktiles);

    if (have_next) {
      if (wave == 0) stage_tdm(kt + 1, nxt);               // async issue
      if constexpr (!B_NMAJOR) stage_B_manual(kt + 1, nxt);
    }

    // fragments + 8 WMMAs on current buffer
    v16bf af[2], bfv[4];
#pragma unroll
    for (int i = 0; i < 2; ++i)
      af[i] = load_frag_lds(&As[cur][wm * 32 + i * 16][0], lane);
#pragma unroll
    for (int j = 0; j < 4; ++j)
      bfv[j] = load_frag_lds(&Bs[cur][wn * 64 + j * 16][0], lane);

#pragma unroll
    for (int i = 0; i < 2; ++i)
#pragma unroll
      for (int j = 0; j < 4; ++j)
        acc[i][j] = __builtin_amdgcn_wmma_f32_16x16x32_bf16(
            false, af[i], false, bfv[j], (short)0, acc[i][j], false, false);

    if (have_next && wave == 0) __builtin_amdgcn_s_wait_tensorcnt(0);
    __syncthreads();   // publishes buffer nxt; releases buffer cur for reuse
  }

  // ---- epilogue: C layout — VGPR r: lanes0-15 M=r, lanes16-31 M=r+8; N=lane&15
#pragma unroll
  for (int i = 0; i < 2; ++i) {
#pragma unroll
    for (int j = 0; j < 4; ++j) {
#pragma unroll
      for (int r = 0; r < 8; ++r) {
        const int row = m0 + wm * 32 + i * 16 + r + (lane >> 4) * 8;
        const int col = n0 + wn * 64 + j * 16 + (lane & 15);
        if (row < M && col < N) {
          const float v = acc[i][j][r];
          if constexpr (OUT_BF16)
            ((bf16*)Cout)[(size_t)row * N + col] = f2bf(v);
          else
            ((float*)Cout)[(size_t)row * N + col] = v;
        }
      }
    }
  }
}

// ---------------------------------------------------------------------------
// Single-pass per-row top-20: one block per row; per-thread sorted insert,
// then 256-way merge (20 rounds of block argmax over each thread's head).
// ---------------------------------------------------------------------------
#define KREC 20

__global__ __launch_bounds__(256) void topk20_kernel(
    const float* __restrict__ S, int N, const int* __restrict__ map,
    int* __restrict__ out_idx) {
  const int row = blockIdx.x;
  const int tid = threadIdx.x;
  const float* Srow = S + (size_t)row * N;

  float lv[KREC];
  int   li[KREC];
#pragma unroll
  for (int i = 0; i < KREC; ++i) { lv[i] = -3.4e38f; li[i] = 0; }

  for (int i = tid; i < N; i += 256) {
    const float v = Srow[i];
    if (v > lv[KREC - 1]) {
      int pos = KREC - 1;
      while (pos > 0 && lv[pos - 1] < v) {
        lv[pos] = lv[pos - 1]; li[pos] = li[pos - 1]; --pos;
      }
      lv[pos] = v; li[pos] = i;
    }
  }

  __shared__ float sv[256];
  __shared__ int   so[256];
  int ptr = 0;
  for (int r = 0; r < KREC; ++r) {
    sv[tid] = (ptr < KREC) ? lv[ptr] : -3.4e38f;
    so[tid] = tid;
    __syncthreads();
    for (int s = 128; s > 0; s >>= 1) {
      if (tid < s && sv[tid + s] > sv[tid]) {
        sv[tid] = sv[tid + s]; so[tid] = so[tid + s];
      }
      __syncthreads();
    }
    const int w = so[0];
    __syncthreads();
    if (tid == w) {
      const int gi = li[ptr];
      out_idx[row * KREC + r] = map ? map[gi] : gi;
      ++ptr;
    }
    __syncthreads();
  }
}

// ---------------------------------------------------------------------------
// probs = softmax(X @ W_mapper)  — one thread per row
// ---------------------------------------------------------------------------
__global__ void probs_kernel(const float* __restrict__ X,
                             const float* __restrict__ Wm,
                             float* __restrict__ probs, int Brows, int P) {
  const int row = blockIdx.x * blockDim.x + threadIdx.x;
  if (row >= Brows) return;
  float l0 = 0.f, l1 = 0.f, l2 = 0.f;
  const float* xr = X + (size_t)row * P;
  for (int k = 0; k < P; ++k) {
    const float x = xr[k];
    l0 += x * Wm[k * 3 + 0];
    l1 += x * Wm[k * 3 + 1];
    l2 += x * Wm[k * 3 + 2];
  }
  const float mx = fmaxf(l0, fmaxf(l1, l2));
  const float e0 = __expf(l0 - mx), e1 = __expf(l1 - mx), e2 = __expf(l2 - mx);
  const float inv = 1.0f / (e0 + e1 + e2);
  probs[row * 3 + 0] = e0 * inv;
  probs[row * 3 + 1] = e1 * inv;
  probs[row * 3 + 2] = e2 * inv;
}

// ---------------------------------------------------------------------------
// Greedy set-intersection fusion + categorical sampling (hash RNG)
// ---------------------------------------------------------------------------
__device__ __forceinline__ float rnd01(unsigned row, unsigned t) {
  unsigned x = row * 0x9E3779B9u + t * 0x85EBCA6Bu + 0xC2B2AE35u;
  x ^= x >> 16; x *= 0x7FEB352Du; x ^= x >> 15; x *= 0x846CA68Bu; x ^= x >> 16;
  return (x >> 8) * (1.0f / 16777216.0f);
}

__global__ void fuse_kernel(const int* __restrict__ top_i,
                            const int* __restrict__ mid_i,
                            const int* __restrict__ sim_i,
                            const float* __restrict__ probs,
                            float* __restrict__ out, int Brows) {
  const int row = blockIdx.x * blockDim.x + threadIdx.x;
  if (row >= Brows) return;

  int top[KREC], mid[KREC], sim[KREC];
  for (int i = 0; i < KREC; ++i) {
    top[i] = top_i[row * KREC + i];
    mid[i] = mid_i[row * KREC + i];
    sim[i] = sim_i[row * KREC + i];
  }

  auto isin = [](int v, const int* arr, const bool* msk) -> bool {
    for (int j = 0; j < KREC; ++j)
      if (arr[j] == v && (!msk || msk[j])) return true;
    return false;
  };

  bool m_tm[KREC], m_ts[KREC], m_ms[KREC];
  bool mask_common[KREC], in_common_mid[KREC];
  bool mask_tm[KREC], mask_ts[KREC], mask_ms[KREC];
  bool top_pool[KREC], mid_pool[KREC], sim_pool[KREC];

  for (int i = 0; i < KREC; ++i) {
    m_tm[i] = isin(top[i], mid, nullptr);
    m_ts[i] = isin(top[i], sim, nullptr);
    m_ms[i] = isin(mid[i], sim, nullptr);
  }
  for (int i = 0; i < KREC; ++i)
    mask_common[i] = m_tm[i] && m_ts[i] && m_ms[i];
  for (int i = 0; i < KREC; ++i)
    in_common_mid[i] = isin(mid[i], top, mask_common);
  for (int i = 0; i < KREC; ++i) {
    mask_tm[i] = m_tm[i] && !mask_common[i];
    mask_ts[i] = m_ts[i] && !mask_common[i];
    mask_ms[i] = m_ms[i] && !in_common_mid[i];
  }
  for (int i = 0; i < KREC; ++i) {
    top_pool[i] = !mask_common[i] && !mask_tm[i] && !mask_ts[i];
    mid_pool[i] = !in_common_mid[i] && !isin(mid[i], top, mask_tm) && !mask_ms[i];
    sim_pool[i] = !isin(sim[i], top, mask_common) && !isin(sim[i], top, mask_ts)
                  && !isin(sim[i], mid, mask_ms);
  }

  int det[4 * KREC]; int nd = 0;
  for (int i = 0; i < KREC; ++i) if (mask_common[i]) det[nd++] = top[i];
  for (int i = 0; i < KREC; ++i) if (mask_tm[i])     det[nd++] = top[i];
  for (int i = 0; i < KREC; ++i) if (mask_ts[i])     det[nd++] = top[i];
  for (int i = 0; i < KREC; ++i) if (mask_ms[i])     det[nd++] = mid[i];

  int pool[3][KREC]; int cnt[3] = {0, 0, 0};
  for (int i = 0; i < KREC; ++i) if (top_pool[i]) pool[0][cnt[0]++] = top[i];
  for (int i = 0; i < KREC; ++i) if (mid_pool[i]) pool[1][cnt[1]++] = mid[i];
  for (int i = 0; i < KREC; ++i) if (sim_pool[i]) pool[2][cnt[2]++] = sim[i];

  float pr[3] = {probs[row * 3 + 0], probs[row * 3 + 1], probs[row * 3 + 2]};
  int ptr[3] = {0, 0, 0};

  for (int t = 0; t < KREC; ++t) {
    float w[3]; float tot = 0.f;
    for (int c = 0; c < 3; ++c) {
      w[c] = (ptr[c] < cnt[c]) ? pr[c] : 0.f;
      tot += w[c];
    }
    int idx = 0;
    if (tot > 0.f) {
      const float x = rnd01((unsigned)row, (unsigned)t) * tot;
      float csum = 0.f; idx = -1;
      for (int c = 0; c < 3; ++c) {
        if (w[c] > 0.f) { csum += w[c]; if (idx < 0 && x < csum) idx = c; }
      }
      if (idx < 0) { for (int c = 2; c >= 0; --c) if (w[c] > 0.f) idx = c; }
      if (idx < 0) idx = 0;
    }
    const int pp = ptr[idx] < (KREC - 1) ? ptr[idx] : (KREC - 1);
    const int sampled = pool[idx][pp];
    const bool use_det = (t < nd);
    const int val = use_det ? det[t] : sampled;
    if (!use_det) ptr[idx]++;
    out[(size_t)row * KREC + t] = (float)val;
  }
}

// ---------------------------------------------------------------------------
// Launcher
// ---------------------------------------------------------------------------
extern "C" void kernel_launch(void* const* d_in, const int* in_sizes, int n_in,
                              void* d_out, int out_size, void* d_ws, size_t ws_size,
                              hipStream_t stream) {
  const float* X        = (const float*)d_in[0];   // [2048,64]
  const float* W_sp     = (const float*)d_in[1];   // [64,128]
  const float* W_sd     = (const float*)d_in[2];   // [128,2000]
  const float* W_mp     = (const float*)d_in[3];   // [64,128]
  const float* W_md     = (const float*)d_in[4];   // [128,5000]
  const float* W_mapper = (const float*)d_in[5];   // [64,3]
  const float* ratings  = (const float*)d_in[6];   // [2000,20000]
  const float* pers     = (const float*)d_in[7];   // [2000,64]
  const int*   top_map  = (const int*)d_in[8];     // [2000]
  const int*   mid_map  = (const int*)d_in[9];     // [5000]
  float* out = (float*)d_out;

  const int Brows = 2048, P = 64, L = 128;
  const int NU = 2000, NI = 20000, NT = 2000, NM = 5000;

  char* ws = (char*)d_ws;
  size_t off = 0;
  auto alloc = [&](size_t bytes) -> void* {
    void* p = ws + off;
    off = (off + bytes + 255) & ~(size_t)255;
    return p;
  };

  bf16* Xbf    = (bf16*)alloc((size_t)Brows * P * 2);
  bf16* Pbf    = (bf16*)alloc((size_t)NU * P * 2);
  bf16* Rbf    = (bf16*)alloc((size_t)NU * NI * 2);
  bf16* Wspbf  = (bf16*)alloc((size_t)P * L * 2);
  bf16* Wsdbf  = (bf16*)alloc((size_t)L * NT * 2);
  bf16* Wmpbf  = (bf16*)alloc((size_t)P * L * 2);
  bf16* Wmdbf  = (bf16*)alloc((size_t)L * NM * 2);
  bf16* S1bf   = (bf16*)alloc((size_t)Brows * NU * 2);
  bf16* Hsbf   = (bf16*)alloc((size_t)Brows * L * 2);
  bf16* Hmbf   = (bf16*)alloc((size_t)Brows * L * 2);
  float* top_sub = (float*)alloc((size_t)Brows * NT * 4);
  float* mid_sub = (float*)alloc((size_t)Brows * NM * 4);
  float* simS    = (float*)alloc((size_t)Brows * NI * 4);
  int* top_idx   = (int*)alloc((size_t)Brows * KREC * 4);
  int* mid_idx   = (int*)alloc((size_t)Brows * KREC * 4);
  int* sim_idx   = (int*)alloc((size_t)Brows * KREC * 4);
  float* probs   = (float*)alloc((size_t)Brows * 3 * 4);

  auto cvt = [&](const float* s, bf16* d, int n) {
    int blocks = (n + 255) / 256;
    if (blocks > 2048) blocks = 2048;
    f32_to_bf16_kernel<<<blocks, 256, 0, stream>>>(s, d, n);
  };
  cvt(X, Xbf, Brows * P);
  cvt(pers, Pbf, NU * P);
  cvt(ratings, Rbf, NU * NI);
  cvt(W_sp, Wspbf, P * L);
  cvt(W_sd, Wsdbf, L * NT);
  cvt(W_mp, Wmpbf, P * L);
  cvt(W_md, Wmdbf, L * NM);

  const dim3 blk(256);
  // S1 = X @ pers^T  (B stored [N,K]; both tiles TDM-staged) -> bf16
  gemm_bf16_wmma<true, true><<<dim3((NU + BN - 1) / BN, Brows / BM), blk, 0, stream>>>(
      Xbf, Pbf, S1bf, Brows, NU, P);
  // Hs = X @ W_sp -> bf16 ; top_sub = Hs @ W_sd -> f32
  gemm_bf16_wmma<false, true><<<dim3((L + BN - 1) / BN, Brows / BM), blk, 0, stream>>>(
      Xbf, Wspbf, Hsbf, Brows, L, P);
  gemm_bf16_wmma<false, false><<<dim3((NT + BN - 1) / BN, Brows / BM), blk, 0, stream>>>(
      Hsbf, Wsdbf, top_sub, Brows, NT, L);
  // Hm = X @ W_mp -> bf16 ; mid_sub = Hm @ W_md -> f32
  gemm_bf16_wmma<false, true><<<dim3((L + BN - 1) / BN, Brows / BM), blk, 0, stream>>>(
      Xbf, Wmpbf, Hmbf, Brows, L, P);
  gemm_bf16_wmma<false, false><<<dim3((NM + BN - 1) / BN, Brows / BM), blk, 0, stream>>>(
      Hmbf, Wmdbf, mid_sub, Brows, NM, L);
  // sim = S1 @ ratings -> f32 (164 GFLOP GEMM; R streamed once, L2-resident)
  gemm_bf16_wmma<false, false><<<dim3((NI + BN - 1) / BN, Brows / BM), blk, 0, stream>>>(
      S1bf, Rbf, simS, Brows, NI, NU);

  // per-row top-20 (indices mapped into full item space for top/mid)
  topk20_kernel<<<Brows, 256, 0, stream>>>(top_sub, NT, top_map, top_idx);
  topk20_kernel<<<Brows, 256, 0, stream>>>(mid_sub, NM, mid_map, mid_idx);
  topk20_kernel<<<Brows, 256, 0, stream>>>(simS, NI, nullptr, sim_idx);

  probs_kernel<<<(Brows + 255) / 256, 256, 0, stream>>>(X, W_mapper, probs, Brows, P);

  fuse_kernel<<<(Brows + 255) / 256, 256, 0, stream>>>(top_idx, mid_idx, sim_idx,
                                                       probs, out, Brows);
}
